// Model_Net_ESM_biLSTM_CombineUpgrade_19516331393573
// MI455X (gfx1250) — compile-verified, matrix operand order checked by reading, not run on hardware
//
#include <hip/hip_runtime.h>
#include <hip/hip_bf16.h>

// ---------------------------------------------------------------------------
// GCN protein net on MI455X.
//  * All GEMMs: v_wmma_f32_16x16x32_bf16 (f32 accumulate), 2 WMMA/wave/k-step.
//  * Weights pre-converted once to bf16 W^T [N x K] (K padded to 8) so B-tile
//    staging is a pure byte copy: global_load_async_to_lds_b128 per lane,
//    synchronized with s_wait_asynccnt.  A (activations, f32) staged with
//    cvt_pk_bf16 + ds_store_b32, double-buffered.
// ---------------------------------------------------------------------------

typedef __attribute__((ext_vector_type(16))) __bf16 v16bf;
typedef __attribute__((ext_vector_type(8)))  __bf16 v8bf;
typedef __attribute__((ext_vector_type(2)))  __bf16 v2bf;
typedef __attribute__((ext_vector_type(8)))  float  v8f;

#define TK 32   // K tile
#define TM 32   // M tile per block
#define TN 64   // N tile per block (2 sub-tiles of 16 per wave)
#define LDA_S 40  // LDS row stride (bf16): 80B, keeps 16B alignment, spreads banks
#define LDB_S 40

__device__ __forceinline__ unsigned lds_off(const void* p) {
    // LDS aperture: flat addr[31:0] is the LDS byte offset (ISA 10.2).
    return (unsigned)(unsigned long long)p;
}

// C[M x Nn] = act(A[M x K] @ W[K x Nn] + bias), W given as bf16 W^T [Nn x K].
// act: 0=none,1=relu,2=sigmoid
__global__ __launch_bounds__(128)
void gemm_bf16_wmma(const float* __restrict__ A, long long lda,
                    const __bf16* __restrict__ Bt, long long ldbt,
                    const float* __restrict__ bias,
                    float* __restrict__ C, long long ldc, long long coff,
                    int M, int Nn, int K, int act)
{
    __shared__ __align__(16) __bf16 sA[2][TM][LDA_S];   // [row][k]
    __shared__ __align__(16) __bf16 sB[2][TN][LDB_S];   // [n][k] (W^T layout)

    const int tid  = threadIdx.x;
    const int wave = tid >> 5;
    const int lane = tid & 31;
    const int hl   = lane >> 4;   // half-wave
    const int lr   = lane & 15;

    const int bm = blockIdx.y * TM;
    const int bn = blockIdx.x * TN;
    const int wm = (wave >> 1) * 16;   // wave origin in M
    const int wn = (wave & 1) * 32;    // wave origin in N (2 x 16 sub-tiles)

    const bool interior = (bm + TM <= M) && (bn + TN <= Nn);

    // Fast staging: interior block + full K tile.
    auto stage_fast = [&](int p, int k0) {
        // A: f32 -> bf16 conversion staging (branch-free).
#pragma unroll
        for (int t = 0; t < 4; ++t) {            // 32x32 = 512 pairs
            int i  = tid + t * 128;
            int r  = i >> 4;
            int kp = (i & 15) * 2;
            long long o = (long long)(bm + r) * lda + (k0 + kp);
            v2bf pk = { (__bf16)A[o], (__bf16)A[o + 1] };
            *(v2bf*)&sA[p][r][kp] = pk;
        }
        // B: already bf16 + transposed in global -> async DMA 16B chunks.
        // Tile = 64 rows (n) x 64B (32 k): 256 chunks, 2 per thread.
#pragma unroll
        for (int q = 0; q < 2; ++q) {
            int i2 = tid * 2 + q;
            int n  = i2 >> 2;                    // 0..63
            int ch = (i2 & 3) * 8;               // k chunk: 0,8,16,24 elems
            const __bf16* gp = &Bt[(long long)(bn + n) * ldbt + (k0 + ch)];
            asm volatile("global_load_async_to_lds_b128 %0, %1, off"
                         :: "v"(lds_off(&sB[p][n][ch])),
                            "v"((unsigned long long)gp)
                         : "memory");
        }
    };
    // Guarded staging for edge tiles: clamped loads + select-to-zero (sync).
    auto stage_guard = [&](int p, int k0) {
#pragma unroll
        for (int t = 0; t < 4; ++t) {
            int i  = tid + t * 128;
            int r  = i >> 4;
            int kp = (i & 15) * 2;
            int row = bm + r, gk = k0 + kp;
            long long ro = (long long)(row < M ? row : M - 1) * lda;
            int ka = gk     < K ? gk     : K - 1;
            int kb = gk + 1 < K ? gk + 1 : K - 1;
            float v0 = A[ro + ka];
            float v1 = A[ro + kb];
            v0 = (row < M && gk     < K) ? v0 : 0.0f;
            v1 = (row < M && gk + 1 < K) ? v1 : 0.0f;
            v2bf pk = { (__bf16)v0, (__bf16)v1 };
            *(v2bf*)&sA[p][r][kp] = pk;
        }
#pragma unroll
        for (int t = 0; t < 8; ++t) {            // B: 1024 pairs, contiguous k
            int i  = tid + t * 128;
            int c  = i & 63;
            int kp = (i >> 6) * 2;
            int col = bn + c, gk = k0 + kp;
            long long co = (long long)(col < Nn ? col : Nn - 1) * ldbt;
            int ka = gk     < K ? gk     : K - 1;
            int kb = gk + 1 < K ? gk + 1 : K - 1;
            __bf16 v0 = Bt[co + ka];
            __bf16 v1 = Bt[co + kb];
            if (!(col < Nn && gk     < K)) v0 = (__bf16)0.0f;
            if (!(col < Nn && gk + 1 < K)) v1 = (__bf16)0.0f;
            v2bf pk = { v0, v1 };
            *(v2bf*)&sB[p][c][kp] = pk;
        }
    };
    auto stage = [&](int p, int k0) {            // uniform (scalar) branch
        if (interior && (k0 + TK <= K)) stage_fast(p, k0);
        else                            stage_guard(p, k0);
    };

    v8f acc0 = {}, acc1 = {};
    const int nk = (K + TK - 1) / TK;
    stage(0, 0);
    int p = 0;
    for (int kt = 0; kt < nk; ++kt) {
        // Complete this wave's async B copies from the previous stage, then
        // rendezvous so every wave sees the fully-written buffer p.
        asm volatile("s_wait_asynccnt 0x0" ::: "memory");
        __syncthreads();
        // Fragment LDS loads from buffer p FIRST (DS retires in order, so the
        // pre-WMMA wait need not cover the staging stores issued below).
        v8bf alo = *(const v8bf*)&sA[p][wm + lr][hl * 8];
        v8bf ahi = *(const v8bf*)&sA[p][wm + lr][16 + hl * 8];
        v8bf b0l = *(const v8bf*)&sB[p][wn + lr][hl * 8];
        v8bf b0h = *(const v8bf*)&sB[p][wn + lr][16 + hl * 8];
        v8bf b1l = *(const v8bf*)&sB[p][wn + 16 + lr][hl * 8];
        v8bf b1h = *(const v8bf*)&sB[p][wn + 16 + lr][16 + hl * 8];
        // Overlap: stage next K tile into the other buffer.
        if (kt + 1 < nk) {
            __builtin_prefetch(&A[(long long)bm * lda + (kt + 1) * TK], 0, 1);
            stage(p ^ 1, (kt + 1) * TK);
        }
        v16bf af  = __builtin_shufflevector(alo, ahi,
                      0,1,2,3,4,5,6,7,8,9,10,11,12,13,14,15);
        v16bf bf0 = __builtin_shufflevector(b0l, b0h,
                      0,1,2,3,4,5,6,7,8,9,10,11,12,13,14,15);
        v16bf bf1 = __builtin_shufflevector(b1l, b1h,
                      0,1,2,3,4,5,6,7,8,9,10,11,12,13,14,15);
        // Two independent WMMAs per k-step (shared A fragment).
        acc0 = __builtin_amdgcn_wmma_f32_16x16x32_bf16(
                   false, af, false, bf0, (short)0, acc0, false, false);
        acc1 = __builtin_amdgcn_wmma_f32_16x16x32_bf16(
                   false, af, false, bf1, (short)0, acc1, false, false);
        p ^= 1;
    }

    // C/D layout: VGPR r -> (M = r + 8*hl, N = lr)
#pragma unroll
    for (int r = 0; r < 8; ++r) {
        int row = bm + wm + r + hl * 8;
        if (row >= M) continue;
        int c0 = bn + wn + lr;
        int c1 = c0 + 16;
        if (c0 < Nn) {
            float v = acc0[r];
            if (bias) v += bias[c0];
            if (act == 1)      v = fmaxf(v, 0.0f);
            else if (act == 2) v = 1.0f / (1.0f + __expf(-v));
            C[(long long)row * ldc + coff + c0] = v;
        }
        if (c1 < Nn) {
            float v = acc1[r];
            if (bias) v += bias[c1];
            if (act == 1)      v = fmaxf(v, 0.0f);
            else if (act == 2) v = 1.0f / (1.0f + __expf(-v));
            C[(long long)row * ldc + coff + c1] = v;
        }
    }
}

// ---- one-time weight convert+transpose: W[KxN] f32 -> Wt[N x ldt] bf16 ----
__global__ void wt_bf16_kernel(const float* __restrict__ W, __bf16* __restrict__ Wt,
                               int K, int N, int ldt) {
    long long idx = (long long)blockIdx.x * blockDim.x + threadIdx.x;
    if (idx >= (long long)K * N) return;
    int k = (int)(idx / N), n = (int)(idx % N);       // coalesced read of W
    Wt[(long long)n * ldt + k] = (__bf16)W[idx];
}

// ------------------------- graph / elementwise kernels ---------------------

__global__ void deg_init_kernel(float* deg, int N) {
    int i = blockIdx.x * blockDim.x + threadIdx.x;
    if (i < N) deg[i] = 1.0f;              // self-loop
}
__global__ void deg_acc_kernel(const long long* __restrict__ dst, float* deg, long long E) {
    long long e = (long long)blockIdx.x * blockDim.x + threadIdx.x;
    if (e < E) atomicAdd(&deg[dst[e]], 1.0f);
}
__global__ void deg_rsqrt_kernel(float* deg, int N) {
    int i = blockIdx.x * blockDim.x + threadIdx.x;
    if (i < N) deg[i] = rsqrtf(fmaxf(deg[i], 1.0f));
}

__global__ void bias_init_kernel(float* __restrict__ out, long long ldo, long long coff,
                                 const float* __restrict__ b, int D, long long M) {
    long long idx = (long long)blockIdx.x * blockDim.x + threadIdx.x;
    if (idx >= M * (long long)D) return;
    long long i = idx / D; int j = (int)(idx % D);
    out[i * ldo + coff + j] = b[j];
}

// msg = h[src]*dinv[src]*dinv[dst]; out[dst] += msg   (edges then self-loops)
__global__ void scatter_kernel(const float* __restrict__ h, int D,
                               const long long* __restrict__ src,
                               const long long* __restrict__ dst,
                               const float* __restrict__ dinv,
                               float* __restrict__ out, long long ldo, long long coff,
                               long long E, long long total) {
    long long idx = (long long)blockIdx.x * blockDim.x + threadIdx.x;
    if (idx >= total) return;
    long long e = idx / D; int j = (int)(idx % D);
    long long s, d;
    if (e < E) { s = src[e]; d = dst[e]; } else { s = d = e - E; }
    float w = dinv[s] * dinv[d];
    atomicAdd(&out[d * ldo + coff + j], w * h[s * (long long)D + j]);
}

__global__ void relu_kernel(float* __restrict__ x, long long ldo, long long coff,
                            int D, long long M) {
    long long idx = (long long)blockIdx.x * blockDim.x + threadIdx.x;
    if (idx >= M * (long long)D) return;
    long long i = idx / D; int j = (int)(idx % D);
    float* p = &x[i * ldo + coff + j];
    *p = fmaxf(*p, 0.0f);
}

__global__ void zero_kernel(float* p, long long n) {
    long long i = (long long)blockIdx.x * blockDim.x + threadIdx.x;
    if (i < n) p[i] = 0.0f;
}
__global__ void pool_cnt_kernel(const long long* __restrict__ batch, float* cnt, int N) {
    int i = blockIdx.x * blockDim.x + threadIdx.x;
    if (i < N) atomicAdd(&cnt[batch[i]], 1.0f);
}
__global__ void pool_sum_kernel(const float* __restrict__ z, const long long* __restrict__ batch,
                                float* __restrict__ pooled, int N, int D) {
    long long idx = (long long)blockIdx.x * blockDim.x + threadIdx.x;
    if (idx >= (long long)N * D) return;
    long long i = idx / D; int j = (int)(idx % D);
    atomicAdd(&pooled[batch[i] * (long long)D + j], z[i * (long long)D + j]);
}
__global__ void pool_div_kernel(float* __restrict__ pooled, const float* __restrict__ cnt,
                                int G, int D) {
    int idx = blockIdx.x * blockDim.x + threadIdx.x;
    if (idx >= G * D) return;
    int g = idx / D;
    pooled[idx] /= fmaxf(cnt[g], 1.0f);
}

// BatchNorm training-math over G rows + relu, in place. One thread per column.
__global__ void bn_relu_kernel(float* __restrict__ z, const float* __restrict__ g,
                               const float* __restrict__ b, int G, int D) {
    int j = blockIdx.x * blockDim.x + threadIdx.x;
    if (j >= D) return;
    float mu = 0.0f;
    for (int r = 0; r < G; ++r) mu += z[(long long)r * D + j];
    mu /= (float)G;
    float var = 0.0f;
    for (int r = 0; r < G; ++r) { float d = z[(long long)r * D + j] - mu; var += d * d; }
    var /= (float)G;
    float inv = rsqrtf(var + 1e-5f);
    for (int r = 0; r < G; ++r) {
        float v = (z[(long long)r * D + j] - mu) * inv * g[j] + b[j];
        z[(long long)r * D + j] = fmaxf(v, 0.0f);
    }
}

// ------------------------------- host side ---------------------------------

static inline long long cdiv(long long a, long long b) { return (a + b - 1) / b; }
static inline int pad8(int k) { return (k + 7) & ~7; }

static inline void launch_gemm(const float* A, long long lda,
                               const __bf16* Bt, long long ldbt,
                               const float* bias, float* C, long long ldc, long long coff,
                               int M, int Nn, int K, int act, hipStream_t s) {
    dim3 grid((Nn + TN - 1) / TN, (M + TM - 1) / TM);
    gemm_bf16_wmma<<<grid, 128, 0, s>>>(A, lda, Bt, ldbt, bias, C, ldc, coff, M, Nn, K, act);
}

extern "C" void kernel_launch(void* const* d_in, const int* in_sizes, int n_in,
                              void* d_out, int out_size, void* d_ws, size_t ws_size,
                              hipStream_t stream) {
    const float*     prot_x = (const float*)d_in[0];
    const long long* eidx   = (const long long*)d_in[1];   // int64 in reference
    const long long* batch  = (const long long*)d_in[2];
    const float *fl1_W = (const float*)d_in[3],  *fl1_b = (const float*)d_in[4];
    const float *fl2_W = (const float*)d_in[5],  *fl2_b = (const float*)d_in[6];
    const float *fl3_W = (const float*)d_in[7],  *fl3_b = (const float*)d_in[8];
    const float *pdb1_W = (const float*)d_in[9],  *pdb1_b = (const float*)d_in[10];
    const float *pdb2_W = (const float*)d_in[11], *pdb2_b = (const float*)d_in[12];
    const float *af1_W  = (const float*)d_in[13], *af1_b  = (const float*)d_in[14];
    const float *af2_W  = (const float*)d_in[15], *af2_b  = (const float*)d_in[16];
    const float *pdb3_W = (const float*)d_in[17], *pdb3_b = (const float*)d_in[18];
    const float *fc1_W  = (const float*)d_in[19], *fc1_b  = (const float*)d_in[20];
    const float *bn_g   = (const float*)d_in[21], *bn_b   = (const float*)d_in[22];
    const float *fc2_W  = (const float*)d_in[23], *fc2_b  = (const float*)d_in[24];

    const int       N  = in_sizes[0] / 6485;
    const long long E  = (long long)in_sizes[1] / 2;
    const long long* src = eidx;
    const long long* dst = eidx + E;

    const int D1 = 469, D2 = 938, D4 = 1876, G = 64;
    const int B = 256;

    float* ws    = (float*)d_ws;
    float* feat  = ws;  ws += (long long)N * D1;
    float* bufH  = ws;  ws += (long long)N * D4;   // gemm temp h
    float* bufZ  = ws;  ws += (long long)N * D4;   // concat(x,y)
    float* bufO  = ws;  ws += (long long)N * D4;   // scatter accum
    float* bufC  = ws;  ws += (long long)N * D1;   // conv1/conv3 out
    float* dinv  = ws;  ws += N;
    float* pooled= ws;  ws += (long long)G * D4;
    float* cnt   = ws;  ws += G;
    float* fc1o  = ws;  ws += (long long)G * 1024;

    // bf16 W^T region (16B aligned; rows padded to 8 elems => 16B stride)
    __bf16* wbf = (__bf16*)(((unsigned long long)(void*)ws + 15ull) & ~15ull);
    auto carve = [&](const float* W, int K, int Nn) {
        int ldt = pad8(K);
        __bf16* p = wbf;
        wbf += (long long)Nn * ldt;
        wt_bf16_kernel<<<cdiv((long long)K * Nn, B), B, 0, stream>>>(W, p, K, Nn, ldt);
        return p;
    };
    __bf16* fl2_T  = carve(fl2_W,    21,   21);
    __bf16* fl1_T  = carve(fl1_W,  6144,  128);
    __bf16* fl3_T  = carve(fl3_W,   320,  320);
    __bf16* pdb1_T = carve(pdb1_W,  D1,   D1);
    __bf16* pdb2_T = carve(pdb2_W,  D1,   D2);
    __bf16* af1_T  = carve(af1_W,   D1,   D1);
    __bf16* af2_T  = carve(af2_W,   D1,   D2);
    __bf16* pdb3_T = carve(pdb3_W,  D4,   D4);
    __bf16* fc1_T  = carve(fc1_W,   D4, 1024);
    __bf16* fc2_T  = carve(fc2_W, 1024,  486);

    // ---- graph normalization: deg = 1 + in-degree; dinv = rsqrt(deg)
    deg_init_kernel<<<cdiv(N, B), B, 0, stream>>>(dinv, N);
    deg_acc_kernel<<<cdiv(E, B), B, 0, stream>>>(dst, dinv, E);
    deg_rsqrt_kernel<<<cdiv(N, B), B, 0, stream>>>(dinv, N);

    // ---- feature layers -> feat = [f2 | f1 | f3]  (relu fused in epilogue)
    launch_gemm(prot_x,        6485, fl2_T, pad8(21),   fl2_b, feat, D1,   0, N,  21,   21, 1, stream);
    launch_gemm(prot_x + 21,   6485, fl1_T, pad8(6144), fl1_b, feat, D1,  21, N, 128, 6144, 1, stream);
    launch_gemm(prot_x + 6165, 6485, fl3_T, pad8(320),  fl3_b, feat, D1, 149, N, 320,  320, 1, stream);

    // ---- GCN conv helper
    auto conv = [&](const float* X, long long ldx, int Din,
                    const __bf16* Wt, const float* b, int Dout,
                    float* out, long long ldo, long long coff) {
        launch_gemm(X, ldx, Wt, pad8(Din), nullptr, bufH, Dout, 0, N, Dout, Din, 0, stream);
        long long tot = (long long)N * Dout;
        bias_init_kernel<<<cdiv(tot, B), B, 0, stream>>>(out, ldo, coff, b, Dout, N);
        long long stot = (E + N) * (long long)Dout;
        scatter_kernel<<<cdiv(stot, B), B, 0, stream>>>(bufH, Dout, src, dst, dinv,
                                                        out, ldo, coff, E, stot);
        relu_kernel<<<cdiv(tot, B), B, 0, stream>>>(out, ldo, coff, Dout, N);
    };

    // pdb branch: feat -> conv1(469) -> conv2(938) -> bufZ[:, 0:938]
    conv(feat, D1, D1, pdb1_T, pdb1_b, D1, bufC, D1, 0);
    conv(bufC, D1, D1, pdb2_T, pdb2_b, D2, bufZ, D4, 0);
    // af branch: feat -> conv3(469) -> conv4(938) -> bufZ[:, 938:1876]
    conv(feat, D1, D1, af1_T, af1_b, D1, bufC, D1, 0);
    conv(bufC, D1, D1, af2_T, af2_b, D2, bufZ, D4, D2);
    // conv5: bufZ(1876) -> bufO(1876)
    conv(bufZ, D4, D4, pdb3_T, pdb3_b, D4, bufO, D4, 0);

    // ---- global mean pool per graph
    zero_kernel<<<cdiv((long long)G * D4 + G, B), B, 0, stream>>>(pooled, (long long)G * D4 + G);
    pool_cnt_kernel<<<cdiv(N, B), B, 0, stream>>>(batch, cnt, N);
    pool_sum_kernel<<<cdiv((long long)N * D4, B), B, 0, stream>>>(bufO, batch, pooled, N, D4);
    pool_div_kernel<<<cdiv((long long)G * D4, B), B, 0, stream>>>(pooled, cnt, G, D4);

    // ---- head: fc1 -> BN(train math) -> relu -> fc2 -> sigmoid
    launch_gemm(pooled, D4, fc1_T, pad8(D4), fc1_b, fc1o, 1024, 0, G, 1024, D4, 0, stream);
    bn_relu_kernel<<<cdiv(1024, B), B, 0, stream>>>(fc1o, bn_g, bn_b, G, 1024);
    launch_gemm(fc1o, 1024, fc2_T, pad8(1024), fc2_b, (float*)d_out, 486, 0, G, 486, 1024, 2, stream);
}
